// ParallelAttention_59459527245934
// MI455X (gfx1250) — compile-verified, matrix operand order checked by reading, not run on hardware
//
#include <hip/hip_runtime.h>
#include <hip/hip_bf16.h>
#include <stdint.h>

#define S_LEN  2048
#define BATCH  4
#define HID    1024
#define NHEAD  16
#define HDIM   64
#define NROWS  (S_LEN * BATCH)   /* 8192 GEMM rows, row = s*B + b */
#define QKV_N  (3 * HID)         /* 3072 */
#define MASK_VAL (-10000.0f)

typedef __bf16 bf16;
typedef __attribute__((ext_vector_type(16))) __bf16 v16bf;
typedef __attribute__((ext_vector_type(8)))  __bf16 v8bf;
typedef __attribute__((ext_vector_type(8)))  float  v8f;
typedef __attribute__((ext_vector_type(4)))  unsigned int v4u;
typedef __attribute__((ext_vector_type(8)))  int v8i_;
typedef __attribute__((ext_vector_type(4)))  int v4i_;

static __device__ inline v8f wmma_bf16(v16bf a, v16bf b, v8f c) {
  // D(16x16 f32) = A(16x32 bf16) * B(32x16 bf16) + C
  return __builtin_amdgcn_wmma_f32_16x16x32_bf16(false, a, false, b, (short)0, c,
                                                 false, false);
}

// A/B fragment loader from a row-major [rows x K] bf16 matrix (global or LDS).
// A-layout (ISA 7.12.2): lanes 0-15 hold row (row0+lane), K packed
// {0..7,16..23}; lanes 16-31 hold same rows, K packed {8..15,24..31}.
// B mirrors A with "row" = N column (load from B^T stored row-major).
static __device__ inline v16bf frag_from_bf16(const bf16* p, int ld, int row0, int k0) {
  const int lane = threadIdx.x & 31;
  const int r  = row0 + (lane & 15);
  const int kb = k0 + ((lane >> 4) << 3);
  const bf16* base = p + (size_t)r * ld + kb;
  union { v16bf v; v8bf h[2]; } u;
  u.h[0] = *(const v8bf*)(base);        // 16B ds/global load
  u.h[1] = *(const v8bf*)(base + 16);   // next K half-block
  return u.v;
}

// ---------------------------------------------------------------------------
// TDM: issue a 2D tensor-tile DMA (global bf16 -> LDS), per ISA ch.8 D#.
// dim0 = contiguous axis (tile_d0 elems/row), dim1 = rows (stride0 elems).
// Tensor dims are set equal to tile dims (tiles never go OOB here).
// ---------------------------------------------------------------------------
static __device__ inline unsigned int lds_off(const void* p) {
  return (unsigned int)(uintptr_t)p;   // LDS aperture: addr[31:0] = LDS offset
}

static __device__ inline void tdm_load_2d(unsigned int lds_addr, const void* gaddr,
                                          unsigned int tile_d0, unsigned int tile_d1,
                                          unsigned int stride0) {
  const unsigned long long ga = (unsigned long long)(uintptr_t)gaddr;
  v4u g0;
  g0[0] = 1u;                                            // count=1, user descriptor
  g0[1] = lds_addr;                                      // lds_addr [63:32]
  g0[2] = (unsigned int)(ga & 0xffffffffu);              // global_addr [95:64]
  g0[3] = (unsigned int)((ga >> 32) & 0x01ffffffu)       // global_addr [120:96]
        | 0x80000000u;                                   // type=2 ("image") [127:126]
  v8i_ g1;
  g1[0] = (int)(1u << 16);                               // data_size=1 (2B), mask=0
  g1[1] = (int)((tile_d0 & 0xffffu) << 16);              // tensor_dim0[15:0] @ [63:48]
  g1[2] = (int)(((tile_d0 >> 16) & 0xffffu)              // tensor_dim0[31:16]
        | ((tile_d1 & 0xffffu) << 16));                  // tensor_dim1[15:0]
  g1[3] = (int)(((tile_d1 >> 16) & 0xffffu)              // tensor_dim1[31:16]
        | ((tile_d0 & 0xffffu) << 16));                  // tile_dim0 @ [127:112]
  g1[4] = (int)(tile_d1 & 0xffffu);                      // tile_dim1 @ [143:128]
  g1[5] = (int)stride0;                                  // tensor_dim0_stride[31:0]
  g1[6] = 0;
  g1[7] = 0;
  const v4i_ z4 = {0, 0, 0, 0};
#if __clang_major__ >= 23
  const v8i_ z8 = {0, 0, 0, 0, 0, 0, 0, 0};
  __builtin_amdgcn_tensor_load_to_lds(g0, g1, z4, z4, z8, 0);
#else
  __builtin_amdgcn_tensor_load_to_lds(g0, g1, z4, z4, 0);
#endif
}

// ---------------------------------------------------------------------------
// Kernel 0: one-shot fp32 -> bf16 conversion (8 elems / thread).
// ---------------------------------------------------------------------------
__global__ __launch_bounds__(256) void cvt_kernel(const float* __restrict__ src,
                                                  bf16* __restrict__ dst, int n) {
  const int i = (blockIdx.x * 256 + threadIdx.x) * 8;
  if (i >= n) return;
  const float4 a = *(const float4*)(src + i);
  const float4 b = *(const float4*)(src + i + 4);
  v8bf o;
  o[0] = (bf16)a.x; o[1] = (bf16)a.y; o[2] = (bf16)a.z; o[3] = (bf16)a.w;
  o[4] = (bf16)b.x; o[5] = (bf16)b.y; o[6] = (bf16)b.z; o[7] = (bf16)b.w;
  *(v8bf*)(dst + i) = o;
}

// ---------------------------------------------------------------------------
// Kernel 1: mixed = hidden @ W_qkv^T + b_qkv -> Q (x 1/8), K, V^T (bf16).
// Block 256 thr = 8 waves (2Mx4N), wave tile 32x64, block tile 64x256.
// A(64x32) and B(256x32) tiles TDM-streamed into double-buffered LDS.
// ---------------------------------------------------------------------------
__global__ __launch_bounds__(256) void qkv_kernel(
    const bf16* __restrict__ X, const bf16* __restrict__ W,
    const float* __restrict__ bias,
    bf16* __restrict__ Q, bf16* __restrict__ K, bf16* __restrict__ VT) {
  __shared__ __align__(16) bf16 Atile[2][64 * 32];
  __shared__ __align__(16) bf16 Btile[2][256 * 32];
  const int wave = threadIdx.x >> 5;
  const int lane = threadIdx.x & 31;
  const int m_blk = blockIdx.y * 64;
  const int n_blk = blockIdx.x * 256;
  const int wm = (wave >> 2) * 32;   // wave M offset in block tile
  const int wn = (wave & 3) * 64;    // wave N offset in block tile

  const v8f zero = {0.f,0.f,0.f,0.f,0.f,0.f,0.f,0.f};
  v8f acc[2][4];
#pragma unroll
  for (int i = 0; i < 2; ++i)
#pragma unroll
    for (int j = 0; j < 4; ++j) acc[i][j] = zero;

  if (threadIdx.x < 32) {            // wave 0 drives the TDM
    tdm_load_2d(lds_off(Atile[0]), X + (size_t)m_blk * HID, 32, 64, HID);
    tdm_load_2d(lds_off(Btile[0]), W + (size_t)n_blk * HID, 32, 256, HID);
  }

  const int NK = HID / 32;
  for (int it = 0; it < NK; ++it) {
    const int buf = it & 1;
    if (threadIdx.x < 32) {
      if (it + 1 < NK) {
        const int kk = (it + 1) * 32;
        tdm_load_2d(lds_off(Atile[buf ^ 1]), X + (size_t)m_blk * HID + kk, 32, 64, HID);
        tdm_load_2d(lds_off(Btile[buf ^ 1]), W + (size_t)n_blk * HID + kk, 32, 256, HID);
        __builtin_amdgcn_s_wait_tensorcnt(2);   // current tile's 2 ops done
      } else {
        __builtin_amdgcn_s_wait_tensorcnt(0);
      }
    }
    __syncthreads();
    const bf16* Ab = Atile[buf];
    const bf16* Bb = Btile[buf];
    v16bf a0 = frag_from_bf16(Ab, 32, wm, 0);
    v16bf a1 = frag_from_bf16(Ab, 32, wm + 16, 0);
#pragma unroll
    for (int j = 0; j < 4; ++j) {
      v16bf bj = frag_from_bf16(Bb, 32, wn + j * 16, 0);
      acc[0][j] = wmma_bf16(a0, bj, acc[0][j]);
      acc[1][j] = wmma_bf16(a1, bj, acc[1][j]);
    }
    __syncthreads();
  }

  const int hb = (lane >> 4) << 3;
  const int nc = lane & 15;
#pragma unroll
  for (int i = 0; i < 2; ++i)
#pragma unroll
    for (int j = 0; j < 4; ++j)
#pragma unroll
      for (int c = 0; c < 8; ++c) {
        const int m = m_blk + wm + i * 16 + hb + c;
        const int n = n_blk + wn + j * 16 + nc;
        float v = acc[i][j][c] + bias[n];
        const int s = m >> 2, b = m & 3;
        const int h = n / 192, r = n % 192;
        const int part = r >> 6, d = r & 63;
        const int bh = b * NHEAD + h;
        if (part == 0) {          // Q, fold in 1/sqrt(64)
          Q[((size_t)bh * S_LEN + s) * HDIM + d] = (bf16)(v * 0.125f);
        } else if (part == 1) {   // K natural (t, d)
          K[((size_t)bh * S_LEN + s) * HDIM + d] = (bf16)v;
        } else {                  // V transposed (d, t)
          VT[((size_t)bh * HDIM + d) * S_LEN + s] = (bf16)v;
        }
      }
}

// ---------------------------------------------------------------------------
// Kernel 2: flash attention per (b,h). Block 128 thr = 4 waves; each wave owns
// 16 queries. K(32x64) and V^T(64x32) tiles TDM-streamed into double-buffered
// LDS shared by all 4 waves; masked online softmax in fp32 via shfl over
// 16-lane halves; P reshaped C-layout -> A-frag through per-wave LDS.
// ---------------------------------------------------------------------------
__global__ __launch_bounds__(128) void attn_kernel(
    const bf16* __restrict__ Q, const bf16* __restrict__ K,
    const bf16* __restrict__ VT, const uint8_t* __restrict__ mask,
    bf16* __restrict__ CTX) {
  __shared__ __align__(16) bf16 Ktile[2][32 * 64];   // rows t, ld 64
  __shared__ __align__(16) bf16 Vtile[2][64 * 32];   // rows d, ld 32
  __shared__ __align__(16) bf16 Plds[4][16][32];
  const int wave = threadIdx.x >> 5;
  const int lane = threadIdx.x & 31;
  const int bh = blockIdx.y;
  const int b  = bh >> 4;
  const int h  = bh & 15;
  const int q0 = blockIdx.x * 64 + wave * 16;
  const bf16* Qh = Q  + (size_t)bh * S_LEN * HDIM;
  const bf16* Kh = K  + (size_t)bh * S_LEN * HDIM;
  const bf16* Vh = VT + (size_t)bh * HDIM * S_LEN;
  const int hb = (lane >> 4) << 3;
  const int nc = lane & 15;

  const v16bf aq0 = frag_from_bf16(Qh, HDIM, q0, 0);    // Q rows, d 0..31
  const v16bf aq1 = frag_from_bf16(Qh, HDIM, q0, 32);   // Q rows, d 32..63

  const v8f zero = {0.f,0.f,0.f,0.f,0.f,0.f,0.f,0.f};
  v8f o[4];
#pragma unroll
  for (int j = 0; j < 4; ++j) o[j] = zero;
  float mrow[8], lrow[8];
  size_t mbase[8];
#pragma unroll
  for (int c = 0; c < 8; ++c) {
    mrow[c] = -1e30f;
    lrow[c] = 0.0f;
    mbase[c] = ((size_t)b * S_LEN + (size_t)(q0 + hb + c)) * S_LEN;
  }

  if (threadIdx.x < 32) {
    tdm_load_2d(lds_off(Ktile[0]), Kh, 64, 32, HDIM);
    tdm_load_2d(lds_off(Vtile[0]), Vh, 32, 64, S_LEN);
  }

  const int NT = S_LEN / 32;
  for (int it = 0; it < NT; ++it) {
    const int buf = it & 1;
    const int t0 = it * 32;
    if (threadIdx.x < 32) {
      if (it + 1 < NT) {
        const int tn = t0 + 32;
        tdm_load_2d(lds_off(Ktile[buf ^ 1]), Kh + (size_t)tn * HDIM, 64, 32, HDIM);
        tdm_load_2d(lds_off(Vtile[buf ^ 1]), Vh + tn, 32, 64, S_LEN);
        __builtin_amdgcn_s_wait_tensorcnt(2);
      } else {
        __builtin_amdgcn_s_wait_tensorcnt(0);
      }
    }
    __syncthreads();
    const bf16* Kb = Ktile[buf];
    const bf16* Vb = Vtile[buf];

    // scores: two 16x16 tiles over keys [t0, t0+32)
    v8f s0 = wmma_bf16(aq0, frag_from_bf16(Kb, HDIM, 0, 0), zero);
    s0     = wmma_bf16(aq1, frag_from_bf16(Kb, HDIM, 0, 32), s0);
    v8f s1 = wmma_bf16(aq0, frag_from_bf16(Kb, HDIM, 16, 0), zero);
    s1     = wmma_bf16(aq1, frag_from_bf16(Kb, HDIM, 16, 32), s1);

    float alpha[8];
#pragma unroll
    for (int c = 0; c < 8; ++c) {
      float x0 = s0[c], x1 = s1[c];
      if (mask[mbase[c] + t0 + nc])      x0 = MASK_VAL;
      if (mask[mbase[c] + t0 + 16 + nc]) x1 = MASK_VAL;
      float mx = fmaxf(x0, x1);
#pragma unroll
      for (int off = 1; off < 16; off <<= 1)
        mx = fmaxf(mx, __shfl_xor(mx, off, 16));
      const float mnew = fmaxf(mrow[c], mx);
      alpha[c] = __expf(mrow[c] - mnew);
      x0 = __expf(x0 - mnew);
      x1 = __expf(x1 - mnew);
      float sum = x0 + x1;
#pragma unroll
      for (int off = 1; off < 16; off <<= 1)
        sum += __shfl_xor(sum, off, 16);
      lrow[c] = lrow[c] * alpha[c] + sum;
      mrow[c] = mnew;
      Plds[wave][hb + c][nc]      = (bf16)x0;   // P row-major (q, t)
      Plds[wave][hb + c][nc + 16] = (bf16)x1;
    }
#pragma unroll
    for (int j = 0; j < 4; ++j)
#pragma unroll
      for (int c = 0; c < 8; ++c) o[j][c] *= alpha[c];

    // P (C-layout) -> A-fragment via per-wave LDS; same-wave LDS is in-order.
    const v16bf pf = frag_from_bf16(&Plds[wave][0][0], 32, 0, 0);
#pragma unroll
    for (int j = 0; j < 4; ++j) {
      v16bf bv = frag_from_bf16(Vb, 32, j * 16, 0);   // V^T tile rows = d cols
      o[j] = wmma_bf16(pf, bv, o[j]);
    }
    __syncthreads();
  }

#pragma unroll
  for (int j = 0; j < 4; ++j)
#pragma unroll
    for (int c = 0; c < 8; ++c) {
      const int srow = q0 + hb + c;
      const int col  = h * HDIM + j * 16 + nc;
      const float v = o[j][c] / lrow[c];
      CTX[(size_t)(srow * BATCH + b) * HID + col] = (bf16)v;
    }
}

// ---------------------------------------------------------------------------
// Kernel 3: out = ctx @ W_dense^T + b_dense (fp32 out). Same TDM-fed GEMM.
// ---------------------------------------------------------------------------
__global__ __launch_bounds__(256) void dense_kernel(
    const bf16* __restrict__ CTX, const bf16* __restrict__ W,
    const float* __restrict__ bias, float* __restrict__ out) {
  __shared__ __align__(16) bf16 Atile[2][64 * 32];
  __shared__ __align__(16) bf16 Btile[2][256 * 32];
  const int wave = threadIdx.x >> 5;
  const int lane = threadIdx.x & 31;
  const int m_blk = blockIdx.y * 64;
  const int n_blk = blockIdx.x * 256;
  const int wm = (wave >> 2) * 32;
  const int wn = (wave & 3) * 64;

  const v8f zero = {0.f,0.f,0.f,0.f,0.f,0.f,0.f,0.f};
  v8f acc[2][4];
#pragma unroll
  for (int i = 0; i < 2; ++i)
#pragma unroll
    for (int j = 0; j < 4; ++j) acc[i][j] = zero;

  if (threadIdx.x < 32) {
    tdm_load_2d(lds_off(Atile[0]), CTX + (size_t)m_blk * HID, 32, 64, HID);
    tdm_load_2d(lds_off(Btile[0]), W + (size_t)n_blk * HID, 32, 256, HID);
  }

  const int NK = HID / 32;
  for (int it = 0; it < NK; ++it) {
    const int buf = it & 1;
    if (threadIdx.x < 32) {
      if (it + 1 < NK) {
        const int kk = (it + 1) * 32;
        tdm_load_2d(lds_off(Atile[buf ^ 1]), CTX + (size_t)m_blk * HID + kk, 32, 64, HID);
        tdm_load_2d(lds_off(Btile[buf ^ 1]), W + (size_t)n_blk * HID + kk, 32, 256, HID);
        __builtin_amdgcn_s_wait_tensorcnt(2);
      } else {
        __builtin_amdgcn_s_wait_tensorcnt(0);
      }
    }
    __syncthreads();
    const bf16* Ab = Atile[buf];
    const bf16* Bb = Btile[buf];
    v16bf a0 = frag_from_bf16(Ab, 32, wm, 0);
    v16bf a1 = frag_from_bf16(Ab, 32, wm + 16, 0);
#pragma unroll
    for (int j = 0; j < 4; ++j) {
      v16bf bj = frag_from_bf16(Bb, 32, wn + j * 16, 0);
      acc[0][j] = wmma_bf16(a0, bj, acc[0][j]);
      acc[1][j] = wmma_bf16(a1, bj, acc[1][j]);
    }
    __syncthreads();
  }

  const int hb = (lane >> 4) << 3;
  const int nc = lane & 15;
#pragma unroll
  for (int i = 0; i < 2; ++i)
#pragma unroll
    for (int j = 0; j < 4; ++j)
#pragma unroll
      for (int c = 0; c < 8; ++c) {
        const int m = m_blk + wm + i * 16 + hb + c;
        const int n = n_blk + wn + j * 16 + nc;
        out[(size_t)m * HID + n] = acc[i][j][c] + bias[n];
      }
}

// ---------------------------------------------------------------------------
extern "C" void kernel_launch(void* const* d_in, const int* in_sizes, int n_in,
                              void* d_out, int out_size, void* d_ws, size_t ws_size,
                              hipStream_t stream) {
  (void)in_sizes; (void)n_in; (void)out_size; (void)ws_size;
  const float*   hidden = (const float*)d_in[0];
  const uint8_t* mask   = (const uint8_t*)d_in[1];  // jnp bool_ = 1 byte
  const float*   Wqkv   = (const float*)d_in[2];
  const float*   bqkv   = (const float*)d_in[3];
  const float*   Wd     = (const float*)d_in[4];
  const float*   bd     = (const float*)d_in[5];
  float* out = (float*)d_out;

  const size_t NX  = (size_t)NROWS * HID;                  // 8,388,608
  const size_t NWQ = (size_t)QKV_N * HID;                  // 3,145,728
  const size_t NWD = (size_t)HID * HID;                    // 1,048,576
  const size_t NQ  = (size_t)BATCH * NHEAD * S_LEN * HDIM; // 8,388,608

  bf16* Xb    = (bf16*)d_ws;      // ~88 MiB total workspace, all bf16
  bf16* Wqb   = Xb   + NX;
  bf16* Wdb   = Wqb  + NWQ;
  bf16* Qb    = Wdb  + NWD;
  bf16* Kb    = Qb   + NQ;
  bf16* VTb   = Kb   + NQ;
  bf16* CTXb  = VTb  + NQ;

  cvt_kernel<<<(int)(NX  / 8 / 256), 256, 0, stream>>>(hidden, Xb,  (int)NX);
  cvt_kernel<<<(int)(NWQ / 8 / 256), 256, 0, stream>>>(Wqkv,   Wqb, (int)NWQ);
  cvt_kernel<<<(int)(NWD / 8 / 256), 256, 0, stream>>>(Wd,     Wdb, (int)NWD);

  qkv_kernel  <<<dim3(QKV_N / 256, NROWS / 64), 256, 0, stream>>>(Xb, Wqb, bqkv,
                                                                  Qb, Kb, VTb);
  attn_kernel <<<dim3(S_LEN / 64, BATCH * NHEAD), 128, 0, stream>>>(Qb, Kb, VTb,
                                                                    mask, CTXb);
  dense_kernel<<<dim3(HID / 256, NROWS / 64), 256, 0, stream>>>(CTXb, Wdb, bd, out);
}
